// QuantLinear_15015205667344
// MI455X (gfx1250) — 8-bit matrix operand order repaired and checked by reading, compiled, not run on hardware
//
#include <hip/hip_runtime.h>

typedef __attribute__((ext_vector_type(16))) int   v16i;
typedef __attribute__((ext_vector_type(8)))  float v8f;
typedef __attribute__((ext_vector_type(4)))  float v4f;
typedef __attribute__((ext_vector_type(4)))  int   v4i;

#define M_DIM 8192
#define N_DIM 16384
#define K_DIM 4096

#define BM 128
#define BN 128
#define BK 128
#define LDST 144
#define NTHREADS 256


__device__ __forceinline__ void async_ld_b128(unsigned lds_byte_off, const unsigned char* gptr) {
    asm volatile("global_load_async_to_lds_b128 %0, %1, off"
                 :: "v"(lds_byte_off), "v"(gptr)
                 : "memory");
}

__device__ __forceinline__ void wait_async0() {
    asm volatile("s_wait_asynccnt 0x0" ::: "memory");
}

__device__ __forceinline__ unsigned lds_offset_of(void* p) {
    typedef __attribute__((address_space(3))) unsigned char* lds_bptr;
    return (unsigned)(unsigned long long)(lds_bptr)p;
}


__device__ __forceinline__ unsigned int f32_to_e4m3_scalar(float f) {
    unsigned int u = __float_as_uint(f);
    unsigned int s = (u >> 24) & 0x80u;
    unsigned int a = u & 0x7FFFFFFFu;
    if (a >= 0x43E00000u) return s | 0x7Eu;
    if (a < 0x3C800000u) {
        float av = __uint_as_float(a);
        int q = (int)rintf(av * 512.0f);
        if (q >= 8) return s | 0x08u;
        return s | (unsigned int)q;
    }
    unsigned int r = a + 0x0007FFFFu + ((a >> 20) & 1u);
    if (r >= 0x43E00000u) return s | 0x7Eu;
    int e = (int)(r >> 23) - 127;
    unsigned int m = (r >> 20) & 7u;
    return s | ((unsigned int)(e + 7) << 3) | m;
}

__device__ __forceinline__ unsigned int pack4_e4m3(float a, float b, float c, float d) {
#if __has_builtin(__builtin_amdgcn_cvt_pk_fp8_f32)
    int v = 0;
    v = __builtin_amdgcn_cvt_pk_fp8_f32(a, b, v, false);
    v = __builtin_amdgcn_cvt_pk_fp8_f32(c, d, v, true);
    return (unsigned int)v;
#else
    return  f32_to_e4m3_scalar(a)
         | (f32_to_e4m3_scalar(b) << 8)
         | (f32_to_e4m3_scalar(c) << 16)
         | (f32_to_e4m3_scalar(d) << 24);
#endif
}

__global__ void quantize_fp8_kernel(const float* __restrict__ in,
                                    unsigned int* __restrict__ out, int n4) {
    int i = blockIdx.x * blockDim.x + threadIdx.x;
    if (i >= n4) return;
    v4f f = __builtin_nontemporal_load(reinterpret_cast<const v4f*>(in) + i);
    float a = fminf(fmaxf(f.x, -448.f), 448.f);
    float b = fminf(fmaxf(f.y, -448.f), 448.f);
    float c = fminf(fmaxf(f.z, -448.f), 448.f);
    float d = fminf(fmaxf(f.w, -448.f), 448.f);
    out[i] = pack4_e4m3(a, b, c, d);
}


__global__ __launch_bounds__(NTHREADS)
void fp8_gemm_wmma_kernel(const unsigned char* __restrict__ Aq,
                          const unsigned char* __restrict__ Bq,
                          const float* __restrict__ bias,
                          float* __restrict__ Cout)
{
    __shared__ unsigned char As[2][BM * LDST];
    __shared__ unsigned char Bs[2][BN * LDST];

    const int tid  = threadIdx.x;
    const int wave = tid >> 5;
    const int lane = tid & 31;
    const int lm   = lane & 15;
    const int lhi  = lane >> 4;

    const int wrow = wave >> 2;
    const int wcol = wave & 3;

    const int gm = blockIdx.y * BM;
    const int gn = blockIdx.x * BN;

    const int grow = tid >> 3;
    const int gcol = (tid & 7) * 16;

    const unsigned char* Abase = Aq + (size_t)gm * K_DIM;
    const unsigned char* Bbase = Bq + (size_t)gn * K_DIM;

    const unsigned AsOff = lds_offset_of(&As[0][0]);
    const unsigned BsOff = lds_offset_of(&Bs[0][0]);

    v8f acc[4][2] = {};

    #pragma unroll
    for (int p = 0; p < 4; ++p) {
        const int r = grow + p * 32;
        async_ld_b128(AsOff + r * LDST + gcol, Abase + (size_t)r * K_DIM + gcol);
        async_ld_b128(BsOff + r * LDST + gcol, Bbase + (size_t)r * K_DIM + gcol);
    }
    wait_async0();
    __syncthreads();

    int buf = 0;
    for (int kc = 0; kc < K_DIM; kc += BK) {
        const int nbuf = buf ^ 1;
        if (kc + BK < K_DIM) {
            #pragma unroll
            for (int p = 0; p < 4; ++p) {
                const int r = grow + p * 32;
                async_ld_b128(AsOff + nbuf * (BM * LDST) + r * LDST + gcol,
                              Abase + (size_t)r * K_DIM + (kc + BK) + gcol);
                async_ld_b128(BsOff + nbuf * (BN * LDST) + r * LDST + gcol,
                              Bbase + (size_t)r * K_DIM + (kc + BK) + gcol);
            }
        }

        v16i afrag[4];
        #pragma unroll
        for (int mt = 0; mt < 4; ++mt) {
            const int row = wrow * 64 + mt * 16 + lm;
            const unsigned char* ap = &As[buf][row * LDST];
            union { v16i v; unsigned long long q[8]; } ua;
            #pragma unroll
            for (int p = 0; p < 8; ++p) {
                const int koff = lhi * 16 + ((p >> 1) * 32) + ((p & 1) * 8);
                ua.q[p] = *reinterpret_cast<const unsigned long long*>(ap + koff);
            }
            afrag[mt] = ua.v;
        }

        v16i bfrag[2];
        #pragma unroll
        for (int nt = 0; nt < 2; ++nt) {
            const int col = wcol * 32 + nt * 16 + lm;
            const unsigned char* bp = &Bs[buf][col * LDST];
            union { v16i v; v4i x[4]; } ub;
            #pragma unroll
            for (int j = 0; j < 4; ++j) {
                const int koff = lhi * 16 + j * 32;
                ub.x[j] = *reinterpret_cast<const v4i*>(bp + koff);
            }
            bfrag[nt] = ub.v;
        }

        #pragma unroll
        for (int mt = 0; mt < 4; ++mt) {
            #pragma unroll
            for (int nt = 0; nt < 2; ++nt) {
                acc[mt][nt] = __builtin_amdgcn_wmma_f32_16x16x128_fp8_fp8(
                    afrag[mt], bfrag[nt], (short)0, acc[mt][nt],
                     false,  false);
            }
        }

        wait_async0();
        __syncthreads();
        buf = nbuf;
    }

    #pragma unroll
    for (int nt = 0; nt < 2; ++nt) {
        const int n = gn + wcol * 32 + nt * 16 + lm;
        const float bv = bias[n];
        #pragma unroll
        for (int mt = 0; mt < 4; ++mt) {
            #pragma unroll
            for (int r = 0; r < 8; ++r) {
                const int m = gm + wrow * 64 + mt * 16 + r + 8 * lhi;
                __builtin_nontemporal_store(acc[mt][nt][r] + bv,
                                            &Cout[(size_t)m * N_DIM + n]);
            }
        }
    }
}


extern "C" void kernel_launch(void* const* d_in, const int* in_sizes, int n_in,
                              void* d_out, int out_size, void* d_ws, size_t ws_size,
                              hipStream_t stream) {
    const float* x    = (const float*)d_in[0];
    const float* W    = (const float*)d_in[1];
    const float* bias = (const float*)d_in[2];
    float* out        = (float*)d_out;

    unsigned char* xq = (unsigned char*)d_ws;
    unsigned char* Wq = xq + (size_t)M_DIM * K_DIM;

    const int nx4 = (M_DIM * K_DIM) / 4;
    const int nw4 = (N_DIM * K_DIM) / 4;
    quantize_fp8_kernel<<<(nx4 + 255) / 256, 256, 0, stream>>>(x, (unsigned int*)xq, nx4);
    quantize_fp8_kernel<<<(nw4 + 255) / 256, 256, 0, stream>>>(W, (unsigned int*)Wq, nw4);

    dim3 grid(N_DIM / BN, M_DIM / BM);
    fp8_gemm_wmma_kernel<<<grid, NTHREADS, 0, stream>>>(xq, Wq, bias, out);
}
